// ScaledDotProductSelector_82789789598321
// MI455X (gfx1250) — matrix operand order repaired and checked by reading, not compiled, not run on hardware
//
#include <hip/hip_runtime.h>

typedef __attribute__((ext_vector_type(16))) __bf16 v16bf;
typedef __attribute__((ext_vector_type(8)))  __bf16 v8bf;
typedef __attribute__((ext_vector_type(4)))  __bf16 v4bf;
typedef __attribute__((ext_vector_type(8)))  float  v8f;

namespace {
constexpr int kBs = 2;      // batch
constexpr int kH  = 16;     // heads
constexpr int kS  = 2048;   // sequence
constexpr int kD  = 128;    // head dim
constexpr int QT  = 128;    // query rows per workgroup (8 waves x 16 rows)
constexpr int KT  = 64;     // keys staged in LDS per tile
constexpr float kNeg = -3.0e38f;

// DPP16 controls
constexpr int kDppQuadXor1 = 0xB1;   // quad_perm(1,0,3,2)
constexpr int kDppQuadXor2 = 0x4E;   // quad_perm(2,3,0,1)
constexpr int kDppHalfMirr = 0x141;  // row_half_mirror (xor 7 within 8)
constexpr int kDppRowMirr  = 0x140;  // row_mirror (xor 15 within 16)
}

template <int CTRL>
__device__ __forceinline__ float dpp_movf(float x) {
  return __int_as_float(__builtin_amdgcn_update_dpp(
      0, __float_as_int(x), CTRL, 0xF, 0xF, true));
}

// Reduce across the 16-lane row group (lanes 0-15 and 16-31 independently),
// entirely on the VALU DPP path -- no LDS, no DScnt waits.
__device__ __forceinline__ float row_sum16(float x) {
  x += dpp_movf<kDppQuadXor1>(x);
  x += dpp_movf<kDppQuadXor2>(x);
  x += dpp_movf<kDppHalfMirr>(x);
  x += dpp_movf<kDppRowMirr>(x);
  return x;
}
__device__ __forceinline__ float row_max16(float x) {
  x = fmaxf(x, dpp_movf<kDppQuadXor1>(x));
  x = fmaxf(x, dpp_movf<kDppQuadXor2>(x));
  x = fmaxf(x, dpp_movf<kDppHalfMirr>(x));
  x = fmaxf(x, dpp_movf<kDppRowMirr>(x));
  return x;
}

__launch_bounds__(256, 1)
__global__ void fa_fwd_causal_wmma(const float* __restrict__ qg,
                                   const float* __restrict__ kg,
                                   const float* __restrict__ vg,
                                   const float* __restrict__ amg,   // [B,1,1,S]
                                   const float* __restrict__ hmg,   // [1,H,1,1]
                                   float* __restrict__ og)
{
  // LDS staging: K natural-major, V transposed, attn_mask slice, P relayout.
  __shared__ __attribute__((aligned(32))) __bf16 Kt[KT][kD];     // 16 KB
  __shared__ __attribute__((aligned(32))) __bf16 Vt[kD][KT];     // 16 KB
  __shared__ __attribute__((aligned(32))) __bf16 Pb[8][16][32];  //  8 KB
  __shared__ __attribute__((aligned(16))) float  Am[KT];         // 256 B

  const int tid  = threadIdx.x;
  const int wave = tid >> 5;
  const int lane = tid & 31;
  const int ln16 = lane & 15;
  const bool hi  = (lane >= 16);

  const int blk = blockIdx.x;
  const int q0  = (blk % (kS / QT)) * QT;
  const int bh  = blk / (kS / QT);       // b*H + h
  const int h   = bh % kH;
  const int b   = bh / kH;

  const size_t base = (size_t)bh * kS * kD;
  const float* qp  = qg + base;
  const float* kp  = kg + base;
  const float* vp  = vg + base;
  const float* amp = amg + (size_t)b * kS;
  const float  headm = hmg[h];

  // ---------------- Q tile -> registers, WMMA A-layout (bf16) ----------------
  const int qrow = q0 + wave * 16 + ln16;
  v16bf qa[4];
  {
    const float* qr = qp + (size_t)qrow * kD;
    #pragma unroll
    for (int c = 0; c < 4; ++c) {
      const int dbase = c * 32 + (hi ? 16 : 0);
      const float4 f0 = *(const float4*)&qr[dbase];
      const float4 f1 = *(const float4*)&qr[dbase + 4];
      const float4 f2 = *(const float4*)&qr[dbase + 8];
      const float4 f3 = *(const float4*)&qr[dbase + 12];
      qa[c][0]  = (__bf16)f0.x; qa[c][1]  = (__bf16)f0.y;
      qa[c][2]  = (__bf16)f0.z; qa[c][3]  = (__bf16)f0.w;
      qa[c][4]  = (__bf16)f1.x; qa[c][5]  = (__bf16)f1.y;
      qa[c][6]  = (__bf16)f1.z; qa[c][7]  = (__bf16)f1.w;
      qa[c][8]  = (__bf16)f2.x; qa[c][9]  = (__bf16)f2.y;
      qa[c][10] = (__bf16)f2.z; qa[c][11] = (__bf16)f2.w;
      qa[c][12] = (__bf16)f3.x; qa[c][13] = (__bf16)f3.y;
      qa[c][14] = (__bf16)f3.z; qa[c][15] = (__bf16)f3.w;
    }
  }

  // ---------------- accumulators (C-layout: VGPR r -> row rbase+r) -----------
  v8f o[8];
  #pragma unroll
  for (int j = 0; j < 8; ++j)
    #pragma unroll
    for (int e = 0; e < 8; ++e) o[j][e] = 0.0f;
  float m[8], l[8];
  #pragma unroll
  for (int r = 0; r < 8; ++r) { m[r] = kNeg; l[r] = 0.0f; }

  const int wrow0 = q0 + wave * 16;    // first query row owned by this wave
  const int kmax  = q0 + QT;           // causal bound (exclusive)

  for (int kb = 0; kb < kmax; kb += KT) {
    __syncthreads();
    // ---- cooperative K/V tile load: f32 -> bf16, V transposed in LDS ----
    {
      const int row  = tid >> 2;          // 0..63
      const int dseg = (tid & 3) * 32;    // 0,32,64,96
      const float* krow = kp + (size_t)(kb + row) * kD + dseg;
      const float* vrow = vp + (size_t)(kb + row) * kD + dseg;
      if (kb + KT < kmax) {               // overlap next tile's HBM latency
        __builtin_prefetch(krow + (size_t)KT * kD, 0, 1);
        __builtin_prefetch(vrow + (size_t)KT * kD, 0, 1);
      }
      if (tid < KT) Am[tid] = amp[kb + tid];
      #pragma unroll
      for (int i = 0; i < 8; ++i) {
        const float4 kf = ((const float4*)krow)[i];
        const float4 vf = ((const float4*)vrow)[i];
        const int d = dseg + i * 4;
        v4bf kpk;
        kpk[0] = (__bf16)kf.x; kpk[1] = (__bf16)kf.y;
        kpk[2] = (__bf16)kf.z; kpk[3] = (__bf16)kf.w;
        *(v4bf*)&Kt[row][d] = kpk;        // one 8-byte LDS store
        Vt[d + 0][row] = (__bf16)vf.x;  Vt[d + 1][row] = (__bf16)vf.y;
        Vt[d + 2][row] = (__bf16)vf.z;  Vt[d + 3][row] = (__bf16)vf.w;
      }
    }
    __syncthreads();

    // ---- two 32-key chunks per tile ----
    for (int ch = 0; ch < 2; ++ch) {
      const int kc    = ch * 32;
      const int kglob = kb + kc;
      if (kglob > wrow0 + 15) continue;   // fully above causal diagonal (wave-uniform)

      // scores: two 16x16 f32 tiles via chained WMMA over D
      v8f c0, c1;
      #pragma unroll
      for (int e = 0; e < 8; ++e) { c0[e] = 0.0f; c1[e] = 0.0f; }
      #pragma unroll
      for (int dc = 0; dc < 4; ++dc) {
        const int doff = dc * 32 + (hi ? 16 : 0);
        v16bf b0 = *(const v16bf*)&Kt[kc + ln16][doff];
        v16bf b1 = *(const v16bf*)&Kt[kc + 16 + ln16][doff];
        c0 = __builtin_amdgcn_wmma_f32_16x16x32_bf16(false, qa[dc], false, b0,
                                                     (short)0, c0, false, false);
        c1 = __builtin_amdgcn_wmma_f32_16x16x32_bf16(false, qa[dc], false, b1,
                                                     (short)0, c1, false, false);
      }

      // additive attn_mask (+ causal mask only on diagonal chunks)
      const float am0 = Am[kc + ln16];
      const float am1 = Am[kc + 16 + ln16];
      const int   rbase = wrow0 + (hi ? 8 : 0);
      float s0v[8], s1v[8];
      if (kglob + 31 <= wrow0) {
        // fast path: entire chunk strictly below diagonal for every row
        #pragma unroll
        for (int r = 0; r < 8; ++r) { s0v[r] = c0[r] + am0; s1v[r] = c1[r] + am1; }
      } else {
        const int col0 = kglob + ln16;
        const int col1 = col0 + 16;
        #pragma unroll
        for (int r = 0; r < 8; ++r) {
          const int row = rbase + r;
          s0v[r] = (col0 <= row) ? (c0[r] + am0) : kNeg;
          s1v[r] = (col1 <= row) ? (c1[r] + am1) : kNeg;
        }
      }

      // online softmax: DPP16 row reductions (VALU-only, co-executes with XDL)
      float pm0[8], pm1[8], scale[8];
      #pragma unroll
      for (int r = 0; r < 8; ++r) {
        const float rm = row_max16(fmaxf(s0v[r], s1v[r]));
        const float mn = fmaxf(m[r], rm);
        const float sc = __expf(m[r] - mn);
        const float p0 = __expf(s0v[r] - mn);
        const float p1 = __expf(s1v[r] - mn);
        const float rs = row_sum16(p0 + p1);
        m[r] = mn;
        l[r] = l[r] * sc + rs;
        scale[r] = sc;
        pm0[r] = p0;
        pm1[r] = p1;
      }

      // C-layout -> A-layout relayout of P via per-wave LDS bounce (bf16)
      #pragma unroll
      for (int r = 0; r < 8; ++r) {
        const int prow = r + (hi ? 8 : 0);
        Pb[wave][prow][ln16]      = (__bf16)pm0[r];
        Pb[wave][prow][16 + ln16] = (__bf16)pm1[r];
      }
      // same-wave cross-lane LDS store->load: enforce ordering explicitly
      asm volatile("s_wait_dscnt 0" ::: "memory");
      v16bf pa;
      {
        const int off = hi ? 16 : 0;
        v8bf t0 = *(const v8bf*)&Pb[wave][ln16][off];
        v8bf t1 = *(const v8bf*)&Pb[wave][ln16][off + 8];
        #pragma unroll
        for (int i = 0; i < 8; ++i) { pa[i] = t0[i]; pa[i + 8] = t1[i]; }
      }

      // rescale running output, then O += P x V (8 WMMAs across D)
      #pragma unroll
      for (int j = 0; j < 8; ++j)
        #pragma unroll
        for (int r = 0; r < 8; ++r) o[j][r] *= scale[r];
      #pragma unroll
      for (int j = 0; j < 8; ++j) {
        v16bf bv = *(const v16bf*)&Vt[j * 16 + ln16][kc + (hi ? 16 : 0)];
        o[j] = __builtin_amdgcn_wmma_f32_16x16x32_bf16(false, pa, false, bv,
                                                       (short)0, o[j], false, false);
      }
    }
  }

  // ---------------- epilogue: normalize, head_mask scale, store --------------
  #pragma unroll
  for (int r = 0; r < 8; ++r) {
    const int row = wrow0 + (hi ? 8 : 0) + r;
    const float inv = headm / l[r];
    float* orow = og + base + (size_t)row * kD;
    #pragma unroll
    for (int j = 0; j < 8; ++j) {
      orow[j * 16 + ln16] = o[j][r] * inv;
    }
  }
}

extern "C" void kernel_launch(void* const* d_in, const int* in_sizes, int n_in,
                              void* d_out, int out_size, void* d_ws, size_t ws_size,
                              hipStream_t stream) {
  (void)in_sizes; (void)n_in; (void)out_size; (void)d_ws; (void)ws_size;
  const float* q  = (const float*)d_in[0];
  const float* k  = (const float*)d_in[1];
  const float* v  = (const float*)d_in[2];
  const float* am = (const float*)d_in[3];
  const float* hm = (const float*)d_in[4];
  float* out = (float*)d_out;

  const int grid = kBs * kH * (kS / QT);   // 512 workgroups
  fa_fwd_causal_wmma<<<grid, 256, 0, stream>>>(q, k, v, am, hm, out);
}
